// HalfKANNUE_68685116998140
// MI455X (gfx1250) — matrix operand order repaired and checked by reading, not compiled, not run on hardware
//
#include <hip/hip_runtime.h>

// Problem constants (match reference)
#define IN_DIM 98304
#define NB     2048
#define H1DIM  256
#define H2DIM  32

// fc1 tiling
#define KSPLIT  8         // grid.y K-slices -> 128*8 = 1024 workgroups
#define KCHUNK  256       // K staged in LDS per iteration (8 WMMA K-steps)
#define FSTRIDE 260       // f32 per LDS row: 256 + 4 pad (TDM pad_amount=4 DW)

typedef __attribute__((ext_vector_type(16))) __bf16 v16bf;
typedef __attribute__((ext_vector_type(8)))  __bf16 v8bf;
typedef __attribute__((ext_vector_type(8)))  float  v8f;
typedef unsigned int u32x4 __attribute__((ext_vector_type(4)));
typedef int          i32x4 __attribute__((ext_vector_type(4)));
typedef int          i32x8 __attribute__((ext_vector_type(8)));

#if __has_builtin(__builtin_amdgcn_tensor_load_to_lds) && \
    __has_builtin(__builtin_amdgcn_s_wait_tensorcnt)
#define HAVE_TDM 1
#else
#define HAVE_TDM 0
#endif

static __device__ __forceinline__ v8bf cvt8(float4 a, float4 b) {
  v8bf r = { (__bf16)a.x, (__bf16)a.y, (__bf16)a.z, (__bf16)a.w,
             (__bf16)b.x, (__bf16)b.y, (__bf16)b.z, (__bf16)b.w };
  return r;
}
static __device__ __forceinline__ float4 relu4(float4 a) {
  a.x = fmaxf(a.x, 0.f); a.y = fmaxf(a.y, 0.f);
  a.z = fmaxf(a.z, 0.f); a.w = fmaxf(a.w, 0.f);
  return a;
}
static __device__ __forceinline__ v16bf ld16f(const float* p) {
  const float4* q = (const float4*)p;
  v8bf lo = cvt8(q[0], q[1]);
  v8bf hi = cvt8(q[2], q[3]);
  return __builtin_shufflevector(lo, hi, 0,1,2,3,4,5,6,7,8,9,10,11,12,13,14,15);
}

#if HAVE_TDM
// LDS offset (bytes) of a __shared__ object: generic -> AS(3) strips aperture.
static __device__ __forceinline__ unsigned int lds_offset_of(const void* p) {
  return (unsigned int)(uintptr_t)(__attribute__((address_space(3))) const void*)p;
}

// Issue one TDM load: x[m0..m0+15, k0..k0+KCHUNK-1] (f32) -> LDS @lds_off,
// padded +4 DWORDs per 256-DWORD row => LDS row stride FSTRIDE=260 f32.
// D# packing per CDNA5 ISA (2D, groups 2/3 zero).
static __device__ __forceinline__ void tdm_issue_x(const float* x, int m0, int k0,
                                                   unsigned int lds_off) {
  unsigned long long ga =
      (unsigned long long)(uintptr_t)(x + (size_t)m0 * IN_DIM + k0);
  u32x4 g0;
  g0[0] = 1u;                                   // count=1, user mode, no gather
  g0[1] = lds_off;                              // lds_addr (bytes)
  g0[2] = (unsigned int)ga;                     // global_addr[31:0]
  g0[3] = (unsigned int)((ga >> 32) & 0x01FFFFFFu) | (2u << 30);  // addr[56:32],type=2
  i32x8 g1;
  g1[0] = (int)((2u << 16)      // data_size = 4 bytes
              | (1u << 20)      // pad_enable
              | (7u << 22)      // pad_interval code 7 = every 256 DWORDs
              | (3u << 25));    // pad_amount  code 3 = 4 DWORDs
  g1[1] = (int)(((unsigned)IN_DIM & 0xFFFFu) << 16);              // dim0[15:0]
  g1[2] = (int)(((unsigned)IN_DIM >> 16) | (((unsigned)NB & 0xFFFFu) << 16)); // dim0[31:16] | dim1[15:0]
  g1[3] = (int)(((unsigned)NB >> 16) | ((unsigned)KCHUNK << 16)); // dim1[31:16] | tile_dim0
  g1[4] = 16;                                                     // tile_dim1=16, tile_dim2=0
  g1[5] = (int)(unsigned)IN_DIM;                                  // dim0_stride[31:0]
  g1[6] = 0;                                                      // dim0_stride[47:32] | dim1_stride[15:0]
  g1[7] = 0;
  i32x4 gz4 = {0, 0, 0, 0};
  i32x8 gz8 = {0, 0, 0, 0, 0, 0, 0, 0};
  // 6-arg form (clang-23 / therock-10.0 headers): groups 2,3 + trailing group,
  // then cache policy.
  __builtin_amdgcn_tensor_load_to_lds(g0, g1, gz4, gz4, gz8, 0);
}
#endif

// ---------------- W1 f32 -> bf16 (once per launch; 48MB, L2-resident) --------
__global__ __launch_bounds__(256) void k_w1cvt(const float* __restrict__ w1,
                                               __bf16* __restrict__ w1b) {
  size_t i = ((size_t)blockIdx.x * 256 + threadIdx.x) * 8;
  const float4* p = (const float4*)(w1 + i);
  *(v8bf*)(w1b + i) = cvt8(p[0], p[1]);
}

// ---------------- h1acc := b1 broadcast (accumulator seed) -------------------
__global__ __launch_bounds__(256) void k_init_h1(float* __restrict__ h1acc,
                                                 const float* __restrict__ b1) {
  h1acc[(size_t)blockIdx.x * H1DIM + threadIdx.x] = b1[threadIdx.x];
}

// ---------------- fc1: x @ W1^T, K-split partials via f32 atomics ------------
// block = 256 threads = 8 waves; owns 16 rows of x (read ONCE from HBM) and
// all 256 output cols (wave w -> n in [32w,32w+32) as two WMMA tiles).
// x chunks staged f32 into double-buffered LDS by the Tensor Data Mover
// (TENSORcnt-tracked, one barrier/chunk) and converted to bf16 per A-fragment.
__global__ __launch_bounds__(256) void k_fc1(const float* __restrict__ x,
                                             const __bf16* __restrict__ w1b,
                                             float* __restrict__ h1acc) {
  __shared__ float sX[2][16 * FSTRIDE];

  const int tid   = threadIdx.x;
  const int lane  = tid & 31;
  const int wv    = tid >> 5;              // 0..7
  const int m0    = blockIdx.x * 16;
  const int kbase = blockIdx.y * (IN_DIM / KSPLIT);

  const int row16 = lane & 15;             // M (A/C) or N (B) index
  const int hi    = lane >> 4;             // K-half select (contiguous K order)
  const int kbA   = hi * 16;               // A frag: lanes 16-31 hold K 16..31
  const int kbB   = hi * 16;               // B frag: lanes 16-31 hold K 16..31
  const int n0    = wv * 32 + row16;       // B tile 0 column
  const int n1    = n0 + 16;               // B tile 1 column

  v8f acc0 = {}; v8f acc1 = {};
  const int nchunks = (IN_DIM / KSPLIT) / KCHUNK;  // 48

#if HAVE_TDM
  if (wv == 0) tdm_issue_x(x, m0, kbase, lds_offset_of(&sX[0][0]));
#else
  const int srow = tid >> 4;               // staging row 0..15
  const int sseg = tid & 15;               // staging 16-float segment
#endif

  for (int c = 0; c < nchunks; ++c) {
    const int k0 = kbase + c * KCHUNK;
    const float* sbuf;
#if HAVE_TDM
    if (wv == 0) __builtin_amdgcn_s_wait_tensorcnt(0);   // chunk c landed
    __syncthreads();                                     // LDS visible; buf (c+1)&1 free
    if (wv == 0 && c + 1 < nchunks)
      tdm_issue_x(x, m0, k0 + KCHUNK, lds_offset_of(&sX[(c + 1) & 1][0]));
    sbuf = &sX[c & 1][0];
#else
    const float* gsrc = x + (size_t)(m0 + srow) * IN_DIM + k0 + sseg * 16;
    const float4* gp = (const float4*)gsrc;
    float4 f0 = gp[0], f1 = gp[1], f2 = gp[2], f3 = gp[3];
    __builtin_prefetch(gsrc + KCHUNK, 0, 3);
    __syncthreads();                                     // prev compute done
    float4* dst = (float4*)&sX[0][srow * FSTRIDE + sseg * 16];
    dst[0] = f0; dst[1] = f1; dst[2] = f2; dst[3] = f3;
    __syncthreads();                                     // staging visible
    sbuf = &sX[0][0];
#endif

#pragma unroll
    for (int kk = 0; kk < KCHUNK / 32; ++kk) {
      // A fragment (16x32 bf16) built from f32 LDS, contiguous K (k = 16*hi + s)
      const float* pa = sbuf + row16 * FSTRIDE + kk * 32 + kbA;
      const float4* qa  = (const float4*)pa;
      const float4* qa2 = (const float4*)(pa + 8);
      v8bf alo = cvt8(qa[0],  qa[1]);        // K kb..kb+7
      v8bf ahi = cvt8(qa2[0], qa2[1]);       // K kb+8..kb+15
      v16bf a  = __builtin_shufflevector(alo, ahi,
                   0,1,2,3,4,5,6,7,8,9,10,11,12,13,14,15);
      // B fragments (32x16 bf16) straight from bf16 W1 (L2-resident)
      const int kg = k0 + kk * 32 + kbB;
      v16bf b0 = *(const v16bf*)(w1b + (size_t)n0 * IN_DIM + kg);
      v16bf b1 = *(const v16bf*)(w1b + (size_t)n1 * IN_DIM + kg);
      acc0 = __builtin_amdgcn_wmma_f32_16x16x32_bf16(false, a, false, b0,
                                                     (short)0, acc0, false, false);
      acc1 = __builtin_amdgcn_wmma_f32_16x16x32_bf16(false, a, false, b1,
                                                     (short)0, acc1, false, false);
    }
  }

  // C layout: lanes0-15 -> M=v, lanes16-31 -> M=v+8; N = (lane&15) + tile base
  const int mrow = m0 + hi * 8;
#pragma unroll
  for (int v = 0; v < 8; ++v) {
    unsafeAtomicAdd(&h1acc[(size_t)(mrow + v) * H1DIM + wv * 32 + row16],      acc0[v]);
    unsafeAtomicAdd(&h1acc[(size_t)(mrow + v) * H1DIM + wv * 32 + 16 + row16], acc1[v]);
  }
}

// ---------------- fused relu(h1) @ W2^T -> relu -> @ Wout^T + bout -----------
__global__ __launch_bounds__(256) void k_fc2_out(const float* __restrict__ h1acc,
                                                 const float* __restrict__ W2,
                                                 const float* __restrict__ b2,
                                                 const float* __restrict__ Wout,
                                                 const float* __restrict__ bout,
                                                 float* __restrict__ out) {
  const int tid  = threadIdx.x;
  const int lane = tid & 31;
  const int wv   = tid >> 5;
  const int m0   = (blockIdx.x * 8 + wv) * 16;
  const int row16 = lane & 15;
  const int hi    = lane >> 4;
  const int kbA   = hi * 16;
  const int kbB   = hi * 16;

  v8f acc0 = {}; v8f acc1 = {};
#pragma unroll
  for (int kk = 0; kk < H1DIM / 32; ++kk) {
    const int k = kk * 32;
    const float* pa = h1acc + (size_t)(m0 + row16) * H1DIM + k + kbA;
    const float4* q  = (const float4*)pa;
    const float4* q2 = (const float4*)(pa + 8);
    v8bf alo = cvt8(relu4(q[0]),  relu4(q[1]));
    v8bf ahi = cvt8(relu4(q2[0]), relu4(q2[1]));
    v16bf a  = __builtin_shufflevector(alo, ahi,
                 0,1,2,3,4,5,6,7,8,9,10,11,12,13,14,15);
    v16bf b0 = ld16f(W2 + (size_t)row16 * H1DIM + k + kbB);
    v16bf b1 = ld16f(W2 + (size_t)(row16 + 16) * H1DIM + k + kbB);
    acc0 = __builtin_amdgcn_wmma_f32_16x16x32_bf16(false, a, false, b0,
                                                   (short)0, acc0, false, false);
    acc1 = __builtin_amdgcn_wmma_f32_16x16x32_bf16(false, a, false, b1,
                                                   (short)0, acc1, false, false);
  }

  const float bb0 = b2[row16], bb1 = b2[row16 + 16];
  const float w0  = Wout[row16], w1 = Wout[row16 + 16];
  const float bo  = bout[0];
#pragma unroll
  for (int v = 0; v < 8; ++v) {
    float s = fmaxf(acc0[v] + bb0, 0.f) * w0 + fmaxf(acc1[v] + bb1, 0.f) * w1;
    s += __shfl_xor(s, 1, 32);
    s += __shfl_xor(s, 2, 32);
    s += __shfl_xor(s, 4, 32);
    s += __shfl_xor(s, 8, 32);
    if (row16 == 0) out[m0 + hi * 8 + v] = s + bo;  // lane0: rows v; lane16: v+8
  }
}

// -----------------------------------------------------------------------------
extern "C" void kernel_launch(void* const* d_in, const int* in_sizes, int n_in,
                              void* d_out, int out_size, void* d_ws, size_t ws_size,
                              hipStream_t stream) {
  const float* x    = (const float*)d_in[0];
  const float* W1   = (const float*)d_in[1];
  const float* b1   = (const float*)d_in[2];
  const float* W2   = (const float*)d_in[3];
  const float* b2   = (const float*)d_in[4];
  const float* Wout = (const float*)d_in[5];
  const float* bout = (const float*)d_in[6];
  float* out = (float*)d_out;

  // workspace: [W1 bf16: 48MB][h1acc f32: 2MB]
  __bf16* w1b   = (__bf16*)d_ws;
  float*  h1acc = (float*)((char*)d_ws + (size_t)H1DIM * IN_DIM * sizeof(__bf16));

  const int cvt_blocks = (int)(((size_t)H1DIM * IN_DIM) / 8 / 256);  // 12288
  k_w1cvt  <<<cvt_blocks,              256, 0, stream>>>(W1, w1b);
  k_init_h1<<<NB,                      256, 0, stream>>>(h1acc, b1);
  k_fc1    <<<dim3(NB / 16, KSPLIT),   256, 0, stream>>>(x, w1b, h1acc);
  k_fc2_out<<<NB / 16 / 8,             256, 0, stream>>>(h1acc, W2, b2, Wout, bout, out);
}
